// StagePolicyNetwork_69355131895813
// MI455X (gfx1250) — matrix operand order repaired and checked by reading, not compiled, not run on hardware
//
#include <hip/hip_runtime.h>

// CDNA5 / gfx1250, wave32. Persistent-block BF16 WMMA (f32 accum) MLP over gathered rows.

typedef __attribute__((ext_vector_type(16))) __bf16 v16bf;
typedef __attribute__((ext_vector_type(8)))  float  v8f;
typedef __attribute__((ext_vector_type(4)))  float  v4f;
typedef __attribute__((ext_vector_type(4)))  unsigned v4u;

#define F_DIM   16
#define E_DIM   64
#define IN_DIM  208     // F + 3*E
#define KPAD    224     // 7 * 32
#define H1      128
#define H2      64
#define NS1     7       // layer-1 K-steps of 32
#define NT1     8       // layer-1 N-tiles of 16
#define NS2     4       // layer-2 K-steps
#define NT2     4       // layer-2 N-tiles
#define WAVES   8
#define BLOCK   256
#define MAXG    16
#define MAXBLK  1024

__device__ __forceinline__ unsigned short f2bf(float f) {
  unsigned u = __builtin_bit_cast(unsigned, f);
  unsigned r = u + 0x7FFFu + ((u >> 16) & 1u);   // round-to-nearest-even
  return (unsigned short)(r >> 16);
}

__global__ __launch_bounds__(BLOCK)
void StagePolicyNetwork_69355131895813_kernel(
    const float* __restrict__ x, const float* __restrict__ h_node,
    const float* __restrict__ h_dag, const float* __restrict__ h_glob,
    const float* __restrict__ W1, const float* __restrict__ b1,
    const float* __restrict__ W2, const float* __restrict__ b2,
    const float* __restrict__ W3, const float* __restrict__ b3,
    const int* __restrict__ stage_idx, const int* __restrict__ batch,
    const int* __restrict__ nsa,
    float* __restrict__ out, int M, int G)
{
  // Fragment-layout LDS (CDNA5 ISA §7.12.2).
  __shared__ __align__(32) unsigned short sA [WAVES * NS1 * 512]; // 57344 B; reused for h1 frags
  __shared__ __align__(32) unsigned short sW1[NS1 * NT1 * 512];   // 57344 B
  __shared__ __align__(32) unsigned short sW2[NS2 * NT2 * 512];   // 16384 B
  __shared__ float sB1[H1];
  __shared__ float sB2[H2];
  __shared__ float sW3[H2];
  __shared__ int   sCum[MAXG + 1];
  __shared__ int   sIdx[128], sBat[128], sG[128];

  const int tid  = threadIdx.x;
  const int wv   = tid >> 5;
  const int lane = tid & 31;

  // ================= one-time staging (amortized over persistent loop) ==========
  if (tid < H1) sB1[tid] = b1[tid];
  if (tid < H2) { sB2[tid] = b2[tid]; sW3[tid] = W3[tid]; }
  if (tid == 0) {
    int acc = 0; sCum[0] = 0;
    int gl = G < MAXG ? G : MAXG;
    for (int g = 0; g < gl; ++g) { acc += nsa[g]; sCum[g + 1] = acc; }
    for (int g = gl; g < MAXG; ++g) sCum[g + 1] = acc;
  }
  // W1 (K padded 208->224) into B-fragment layout: lane = n%16 + 16*((kk/8)%2), e = kk%8 + 8*(kk/16)
  for (int i = tid; i < KPAD * H1; i += BLOCK) {
    int k = i >> 7, n = i & 127;
    float v = (k < IN_DIM) ? W1[k * H1 + n] : 0.f;
    int s = k >> 5, kk = k & 31, t = n >> 4, nn = n & 15;
    int ln = nn + (((kk >> 3) & 1) << 4);
    sW1[(((s << 3) + t) << 9) + (ln << 4) + ((kk & 7) + (((kk >> 4) & 1) << 3))] = f2bf(v);
  }
  // W2 into B-fragment layout
  for (int i = tid; i < H1 * H2; i += BLOCK) {
    int k = i >> 6, n = i & 63;
    float v = W2[k * H2 + n];
    int s = k >> 5, kk = k & 31, t = n >> 4, nn = n & 15;
    int ln = nn + (((kk >> 3) & 1) << 4);
    sW2[(((s << 2) + t) << 9) + (ln << 4) + ((kk & 7) + (((kk >> 4) & 1) << 3))] = f2bf(v);
  }

  const int nBat   = (M + 127) >> 7;
  const int colLo  = lane & 15;
  const int rowOff = (lane >> 4) << 3;   // C layout: VGPR j <-> rows j / j+8
  const v8f vzero  = {0.f, 0.f, 0.f, 0.f, 0.f, 0.f, 0.f, 0.f};

  // ================= persistent loop over 128-row batches =======================
  for (int bat = blockIdx.x; bat < nBat; bat += gridDim.x) {
    __syncthreads();   // weights ready (iter 0) / previous compute done (later iters)

    // ---- per-row metadata: gather index, dag id, graph id (repeat_interleave) ----
    if (tid < 128) {
      int m = bat * 128 + tid;
      int si = -1, bb = 0, gg = 0;
      if (m < M) {
        si = stage_idx[m];
        bb = batch[si];
        for (int g = 0; g < MAXG; ++g) if (m >= sCum[g + 1]) gg = g + 1;
        if (gg > MAXG - 1) gg = MAXG - 1;
      }
      sIdx[tid] = si; sBat[tid] = bb; sG[tid] = gg;
    }
    __syncthreads();

    // ---- gather + concat + bf16 convert, 8 columns per unit ----
    // Aligned 8-col runs never cross a segment boundary (16/80/144/208) and map to
    // 8 contiguous bf16 of one lane's A-fragment -> 2x global b128 + 1x ds_store_b128.
    for (int i = tid; i < 128 * (KPAD / 8); i += BLOCK) {
      int r    = i / (KPAD / 8);
      int cs   = i - r * (KPAD / 8);
      int col0 = cs << 3;
      int si   = sIdx[r];
      v4f lo = {0.f, 0.f, 0.f, 0.f}, hi = {0.f, 0.f, 0.f, 0.f};
      if (si >= 0 && col0 < IN_DIM) {
        if (col0 < 80) {              // x / h_node: streamed once -> non-temporal
          const v4f* p = (col0 < F_DIM)
                       ? (const v4f*)(x + si * F_DIM + col0)
                       : (const v4f*)(h_node + si * E_DIM + (col0 - F_DIM));
          lo = __builtin_nontemporal_load(p);
          hi = __builtin_nontemporal_load(p + 1);
        } else {                      // h_dag / h_glob: heavily reused -> cached
          const v4f* p = (col0 < 144)
                       ? (const v4f*)(h_dag + sBat[r] * E_DIM + (col0 - 80))
                       : (const v4f*)(h_glob + sG[r] * E_DIM + (col0 - 144));
          lo = *p;
          hi = *(p + 1);
        }
      }
      v4u pk;
      pk[0] = (unsigned)f2bf(lo[0]) | ((unsigned)f2bf(lo[1]) << 16);
      pk[1] = (unsigned)f2bf(lo[2]) | ((unsigned)f2bf(lo[3]) << 16);
      pk[2] = (unsigned)f2bf(hi[0]) | ((unsigned)f2bf(hi[1]) << 16);
      pk[3] = (unsigned)f2bf(hi[2]) | ((unsigned)f2bf(hi[3]) << 16);
      int w = r >> 4, rr = r & 15;
      int s = col0 >> 5, kk = col0 & 31;
      int ln = rr + (((kk >> 3) & 1) << 4);
      int e0 = ((kk >> 4) & 1) << 3;
      *reinterpret_cast<v4u*>(&sA[((w * NS1 + s) << 9) + (ln << 4) + e0]) = pk;
    }
    __syncthreads();

    const int chunk = bat * WAVES + wv;
    if (chunk * 16 < M) {   // wave-uniform: EXEC all-ones for WMMA
      // ---- layer 1: [16,224] x [224,128] ----
      // K-step outer; batch-preload all 8 B fragments (deep LDS pipeline), then run
      // the 8 WMMAs back-to-back (shared A, distinct D -> no hazards, no waits).
      v8f acc[NT1];
#pragma unroll
      for (int t = 0; t < NT1; ++t) acc[t] = vzero;
#pragma unroll
      for (int s = 0; s < NS1; ++s) {
        v16bf a = *reinterpret_cast<const v16bf*>(&sA[((wv * NS1 + s) << 9) + (lane << 4)]);
        v16bf bfr[NT1];
#pragma unroll
        for (int t = 0; t < NT1; ++t)
          bfr[t] = *reinterpret_cast<const v16bf*>(&sW1[(((s << 3) + t) << 9) + (lane << 4)]);
#pragma unroll
        for (int t = 0; t < NT1; ++t)
          acc[t] = __builtin_amdgcn_wmma_f32_16x16x32_bf16(false, a, false, bfr[t],
                                                           (short)0, acc[t], false, false);
      }
      // bias + relu, write h1 back into this wave's (now dead) sA region as layer-2 A-frags
#pragma unroll
      for (int t = 0; t < NT1; ++t) {
        int col = (t << 4) + colLo;
        float bias = sB1[col];
        int s2 = col >> 5, kk = col & 31;
        int hb = (kk >> 3) & 1;
        int e2 = (kk & 7) + (((kk >> 4) & 1) << 3);
#pragma unroll
        for (int j = 0; j < 8; ++j) {
          float v = acc[t][j] + bias;
          v = v > 0.f ? v : 0.f;
          int ln2 = (j + rowOff) + (hb << 4);
          sA[((wv * NS1 + s2) << 9) + (ln2 << 4) + e2] = f2bf(v);
        }
      }

      // ---- layer 2: [16,128] x [128,64] ----
      v8f acc2[NT2];
#pragma unroll
      for (int t = 0; t < NT2; ++t) acc2[t] = vzero;
#pragma unroll
      for (int s = 0; s < NS2; ++s) {
        v16bf a = *reinterpret_cast<const v16bf*>(&sA[((wv * NS1 + s) << 9) + (lane << 4)]);
        v16bf bfr[NT2];
#pragma unroll
        for (int t = 0; t < NT2; ++t)
          bfr[t] = *reinterpret_cast<const v16bf*>(&sW2[(((s << 2) + t) << 9) + (lane << 4)]);
#pragma unroll
        for (int t = 0; t < NT2; ++t)
          acc2[t] = __builtin_amdgcn_wmma_f32_16x16x32_bf16(false, a, false, bfr[t],
                                                            (short)0, acc2[t], false, false);
      }

      // ---- layer 3: relu(h2) . W3 via VALU + wave32 xor-shuffle reduction ----
      float pl[8];
#pragma unroll
      for (int j = 0; j < 8; ++j) pl[j] = 0.f;
#pragma unroll
      for (int t = 0; t < NT2; ++t) {
        int col = (t << 4) + colLo;
        float bias = sB2[col];
        float w3v  = sW3[col];
#pragma unroll
        for (int j = 0; j < 8; ++j) {
          float v = acc2[t][j] + bias;
          v = v > 0.f ? v : 0.f;
          pl[j] += v * w3v;
        }
      }
#pragma unroll
      for (int msk = 1; msk <= 8; msk <<= 1) {
#pragma unroll
        for (int j = 0; j < 8; ++j)
          pl[j] += __shfl_xor(pl[j], msk, 32);
      }

      float b3v = b3[0];
      int r0 = chunk << 4;
      if (lane == 0) {
#pragma unroll
        for (int j = 0; j < 8; ++j) { int row = r0 + j;     if (row < M) out[row] = pl[j] + b3v; }
      } else if (lane == 16) {
#pragma unroll
        for (int j = 0; j < 8; ++j) { int row = r0 + 8 + j; if (row < M) out[row] = pl[j] + b3v; }
      }
    }
  }
}

extern "C" void kernel_launch(void* const* d_in, const int* in_sizes, int n_in,
                              void* d_out, int out_size, void* d_ws, size_t ws_size,
                              hipStream_t stream) {
  const float* x      = (const float*)d_in[0];
  const float* h_node = (const float*)d_in[1];
  const float* h_dag  = (const float*)d_in[2];
  const float* h_glob = (const float*)d_in[3];
  const float* W1     = (const float*)d_in[4];
  const float* b1     = (const float*)d_in[5];
  const float* W2     = (const float*)d_in[6];
  const float* b2     = (const float*)d_in[7];
  const float* W3     = (const float*)d_in[8];
  const float* b3     = (const float*)d_in[9];
  const int* stage_idx = (const int*)d_in[10];
  const int* batch     = (const int*)d_in[11];
  const int* nsa       = (const int*)d_in[12];

  int M = in_sizes[10];          // number of masked rows
  int G = in_sizes[12];          // number of graphs
  int nBat = (M + 127) / 128;
  int blocks = nBat < MAXBLK ? nBat : MAXBLK;

  StagePolicyNetwork_69355131895813_kernel<<<blocks, BLOCK, 0, stream>>>(
      x, h_node, h_dag, h_glob, W1, b1, W2, b2, W3, b3,
      stage_idx, batch, nsa, (float*)d_out, M, G);
}
